// EEModel_18056042512815
// MI455X (gfx1250) — matrix operand order repaired and checked by reading, not compiled, not run on hardware
//
#include <hip/hip_runtime.h>
#include <hip/hip_bf16.h>
#include <math.h>

typedef __attribute__((ext_vector_type(16))) _Float16 v16h;
typedef __attribute__((ext_vector_type(8)))  _Float16 v8h;
typedef __attribute__((ext_vector_type(8)))  float    v8f;

#define TPB 256

// ---------------- device helpers ----------------

__device__ __forceinline__ float ftanh_f(float x) {
    float e = __expf(2.f * x);            // saturates correctly: e->inf => 1, e->0 => -1
    return 1.f - 2.f / (e + 1.f);
}
__device__ __forceinline__ float sigmoid_f(float x) { return 1.f / (1.f + __expf(-x)); }
__device__ __forceinline__ float gelu_f(float x) {
    float t = ftanh_f(0.7978845608028654f * (x + 0.044715f * x * x * x));
    return 0.5f * x * (1.f + t);
}
__device__ __forceinline__ float lrelu_f(float x) { return x > 0.f ? x : 0.2f * x; }

// monotonic float<->uint encoding for atomicMax on floats
__device__ __forceinline__ unsigned ford(float f) {
    unsigned u = __float_as_uint(f);
    return (u & 0x80000000u) ? ~u : (u | 0x80000000u);
}
__device__ __forceinline__ float finv(unsigned u) {
    return (u & 0x80000000u) ? __uint_as_float(u ^ 0x80000000u) : __uint_as_float(~u);
}

// A fragment (f16 16x32 MxK). lane L: M=L%16; K(e) = 8*(L/16) + (e<8 ? e : e+8)
// -> two contiguous 16B runs. NO bounds checks: callers guarantee padded rows.
__device__ __forceinline__ v16h load_a(const _Float16* __restrict__ A, int lda,
                                       int m0, int k0, int lane) {
    int g = lane >> 4, mr = lane & 15;
    const _Float16* p = A + (size_t)(m0 + mr) * lda + k0 + 8 * g;
    union { v16h v; v8h h[2]; } u;
    u.h[0] = *(const v8h*)p;
    u.h[1] = *(const v8h*)(p + 16);
    return u.v;
}
// B fragment from B stored [N,K]. lane L: N=L%16; K(e) = 8*(L/16) + (e<8 ? e : e+8) -> two 16B runs.
__device__ __forceinline__ v16h load_b(const _Float16* __restrict__ B, int ldb,
                                       int n0, int k0, int lane) {
    int g = lane >> 4, nr = lane & 15;
    const _Float16* p = B + (size_t)(n0 + nr) * ldb + k0 + 8 * g;
    union { v16h v; v8h h[2]; } u;
    u.h[0] = *(const v8h*)p;
    u.h[1] = *(const v8h*)(p + 16);
    return u.v;
}
// A fragment built from f32 source row block (padded to 16 rows)
__device__ __forceinline__ v16h load_a_f32(const float* __restrict__ A, int lda,
                                           int k0, int lane) {
    int g = lane >> 4, mr = lane & 15;
    const float* p = A + (size_t)mr * lda + k0 + 8 * g;
    v8f lo = *(const v8f*)p;
    v8f hi = *(const v8f*)(p + 16);
    v16h a;
#pragma unroll
    for (int e = 0; e < 8; ++e) { a[e] = (_Float16)lo[e]; a[e + 8] = (_Float16)hi[e]; }
    return a;
}

// ---------------- utility kernels ----------------

__global__ void zero_f(float* p, int n) {
    int t = blockIdx.x * blockDim.x + threadIdx.x;
    if (t < n) p[t] = 0.f;
}
__global__ void f2h(const float* s, _Float16* d, int n) {
    int t = blockIdx.x * blockDim.x + threadIdx.x;
    if (t < n) d[t] = (_Float16)s[t];
}
// transpose-convert: src [K,N] f32 -> dst [N,K] f16
__global__ void f2h_t(const float* s, _Float16* d, int K, int N) {
    int t = blockIdx.x * blockDim.x + threadIdx.x;
    if (t >= K * N) return;
    int k = t / N, n = t % N;
    d[(size_t)n * K + k] = (_Float16)s[t];
}

// x[B,129,800] = concat(bert_out, pos_emb), pos row 0 zeroed
__global__ void build_x(const float* bert, const int* pos_ids, const float* pos_table,
                        float* xf, _Float16* xh) {
    int t = blockIdx.x * blockDim.x + threadIdx.x;
    if (t >= 516 * 800) return;
    int c = t % 800, r = t / 800;
    int b = r / 129, tt = r % 129;
    float v;
    if (c < 768) v = bert[(size_t)r * 768 + c];
    else v = (tt == 0) ? 0.f : pos_table[pos_ids[b * 128 + tt - 1] * 32 + (c - 768)];
    xf[t] = v; xh[t] = (_Float16)v;
}

// fused head weights, TRANSPOSED [320,800] = rows [trig(66)|rs(122)|re(122)|pad(10)]
__global__ void build_heads(const float* trigW, const float* trigb,
                            const float* rsW, const float* rsb,
                            const float* reW, const float* reb,
                            _Float16* Wt, float* bh) {
    int t = blockIdx.x * blockDim.x + threadIdx.x;
    if (t >= 320 * 800) return;
    int n = t / 800, k = t % 800;
    float v = (n < 66) ? trigW[k * 66 + n]
            : (n < 188) ? rsW[k * 122 + (n - 66)]
            : (n < 310) ? reW[k * 122 + (n - 188)] : 0.f;
    Wt[t] = (_Float16)v;
    if (k == 0) {
        float bb = (n < 66) ? trigb[n] : (n < 188) ? rsb[n - 66]
                 : (n < 310) ? reb[n - 188] : 0.f;
        bh[n] = bb;
    }
}

// ------------- generic WMMA GEMM, 32x32 tile per wave --------------------------
// C[M,N] f32 = A[M,K] f16 * B[N,K]^T f16 (+bias). K%32==0, N%32==0.
// A rows padded: loads are unguarded; stores guarded by M.
__global__ void gemm_f16(const _Float16* __restrict__ A, const _Float16* __restrict__ B,
                         const float* bias, float* __restrict__ C,
                         int M, int K, int lda, int ldb, int ldc) {
    int lane = threadIdx.x;
    int n0 = blockIdx.x * 32, m0 = blockIdx.y * 32;
    v8f c00 = {}, c01 = {}, c10 = {}, c11 = {};
    for (int k0 = 0; k0 < K; k0 += 32) {
        v16h a0 = load_a(A, lda, m0, k0, lane);
        v16h a1 = load_a(A, lda, m0 + 16, k0, lane);
        v16h b0 = load_b(B, ldb, n0, k0, lane);
        v16h b1 = load_b(B, ldb, n0 + 16, k0, lane);
        c00 = __builtin_amdgcn_wmma_f32_16x16x32_f16(false, a0, false, b0, (short)0, c00, false, false);
        c01 = __builtin_amdgcn_wmma_f32_16x16x32_f16(false, a0, false, b1, (short)0, c01, false, false);
        c10 = __builtin_amdgcn_wmma_f32_16x16x32_f16(false, a1, false, b0, (short)0, c10, false, false);
        c11 = __builtin_amdgcn_wmma_f32_16x16x32_f16(false, a1, false, b1, (short)0, c11, false, false);
    }
    int g = lane >> 4, nr = lane & 15;
    int na = n0 + nr, nb = n0 + 16 + nr;
    float bva = bias ? bias[na] : 0.f;
    float bvb = bias ? bias[nb] : 0.f;
#pragma unroll
    for (int v = 0; v < 8; ++v) {
        int m = m0 + v + 8 * g;
        if (m < M) {
            C[(size_t)m * ldc + na] = c00[v] + bva;
            C[(size_t)m * ldc + nb] = c01[v] + bvb;
        }
        int m2 = m + 16;
        if (m2 < M) {
            C[(size_t)m2 * ldc + na] = c10[v] + bva;
            C[(size_t)m2 * ldc + nb] = c11[v] + bvb;
        }
    }
}

// ---------------- LSTM step ----------------
// gates = Xproj(row) + h@Whh^T ; one wave per 16-wide hidden slice computes all
// four gate tiles so c/h update is lane-local. hprev padded to 16 rows.
__global__ void lstm_step(const float* __restrict__ Xproj, const _Float16* __restrict__ Whh,
                          const float* __restrict__ hprev, float* __restrict__ hnext,
                          float* __restrict__ c, _Float16* __restrict__ lstm_out,
                          int t, int dcol) {
    const int Hd = 800, T = 129;
    int lane = threadIdx.x;
    int n0 = blockIdx.x * 16;
    int g = lane >> 4, lr = lane & 15;
    v8f a0 = {}, a1 = {}, a2 = {}, a3 = {};
    for (int k0 = 0; k0 < Hd; k0 += 32) {
        v16h a = load_a_f32(hprev, Hd, k0, lane);
        v16h b0 = load_b(Whh, Hd, 0 * Hd + n0, k0, lane);
        v16h b1 = load_b(Whh, Hd, 1 * Hd + n0, k0, lane);
        v16h b2 = load_b(Whh, Hd, 2 * Hd + n0, k0, lane);
        v16h b3 = load_b(Whh, Hd, 3 * Hd + n0, k0, lane);
        a0 = __builtin_amdgcn_wmma_f32_16x16x32_f16(false, a, false, b0, (short)0, a0, false, false);
        a1 = __builtin_amdgcn_wmma_f32_16x16x32_f16(false, a, false, b1, (short)0, a1, false, false);
        a2 = __builtin_amdgcn_wmma_f32_16x16x32_f16(false, a, false, b2, (short)0, a2, false, false);
        a3 = __builtin_amdgcn_wmma_f32_16x16x32_f16(false, a, false, b3, (short)0, a3, false, false);
    }
    int n = n0 + lr;
#pragma unroll
    for (int v = 0; v < 8; ++v) {
        int m = v + 8 * g;
        if (m < 4) {
            int r = m * T + t;
            const float* xr = Xproj + (size_t)r * 3200;
            float gi = a0[v] + xr[n];
            float gf = a1[v] + xr[800 + n];
            float gg = a2[v] + xr[1600 + n];
            float go = a3[v] + xr[2400 + n];
            float cn = sigmoid_f(gf) * c[m * Hd + n] + sigmoid_f(gi) * ftanh_f(gg);
            c[m * Hd + n] = cn;
            float h = sigmoid_f(go) * ftanh_f(cn);
            hnext[m * Hd + n] = h;
            lstm_out[(size_t)r * 1600 + dcol + n] = (_Float16)h;
        }
    }
}

// x_g[b,s] = x[b,s+1] + relu(fcout[b,s+1])  (drop CLS row)
__global__ void fc_combine(const float* xf, const float* fcout, float* xg, _Float16* xgh) {
    int t = blockIdx.x * blockDim.x + threadIdx.x;
    if (t >= 512 * 800) return;
    int c = t % 800, r = t / 800;
    int b = r / 128, s = r % 128;
    size_t sr = (size_t)(b * 129 + s + 1) * 800 + c;
    float v = fcout[sr]; v = v > 0.f ? v : 0.f;
    float o = xf[sr] + v;
    xg[t] = o; xgh[t] = (_Float16)o;
}

// ---------------- GAT attention kernels ----------------
__device__ __forceinline__ void edge_sd(const int* esrc, const int* edst,
                                        int b, int e, int& s, int& d) {
    if (e < 256) { s = esrc[b * 256 + e]; d = edst[b * 256 + e]; }
    else { s = e - 256; d = e - 256; }
}

__global__ void att_prep(const float* Hg, const float* asrc, const float* adst,
                         float* a_s, float* a_d, unsigned* menc, float* den) {
    int t = blockIdx.x * blockDim.x + threadIdx.x;
    if (t >= 512 * 8) return;
    int node = t / 8, hd = t % 8;
    float s = 0.f, d = 0.f;
    for (int k = 0; k < 100; ++k) {
        float h = Hg[(size_t)node * 800 + hd * 100 + k];
        s += h * asrc[hd * 100 + k];
        d += h * adst[hd * 100 + k];
    }
    a_s[t] = s; a_d[t] = d;
    menc[t] = ford(-INFINITY); den[t] = 0.f;
}

__global__ void edge_max(const int* esrc, const int* edst,
                         const float* a_s, const float* a_d, unsigned* menc) {
    int t = blockIdx.x * blockDim.x + threadIdx.x;
    if (t >= 4 * 384 * 8) return;
    int b = t / (384 * 8), r = t % (384 * 8), e = r / 8, hd = r % 8;
    int s, d; edge_sd(esrc, edst, b, e, s, d);
    int ns = b * 128 + s, nd = b * 128 + d;
    float v = lrelu_f(a_s[ns * 8 + hd] + a_d[nd * 8 + hd]);
    atomicMax(menc + nd * 8 + hd, ford(v));
}

__global__ void edge_exp(const int* esrc, const int* edst,
                         const float* a_s, const float* a_d, const unsigned* menc,
                         float* den, float* exbuf) {
    int t = blockIdx.x * blockDim.x + threadIdx.x;
    if (t >= 4 * 384 * 8) return;
    int b = t / (384 * 8), r = t % (384 * 8), e = r / 8, hd = r % 8;
    int s, d; edge_sd(esrc, edst, b, e, s, d);
    int ns = b * 128 + s, nd = b * 128 + d;
    float v = lrelu_f(a_s[ns * 8 + hd] + a_d[nd * 8 + hd]);
    float ex = expf(v - finv(menc[nd * 8 + hd]));
    exbuf[t] = ex;
    atomicAdd(den + nd * 8 + hd, ex);
}

__global__ void edge_aggr(const int* esrc, const int* edst, const float* Hg,
                          const float* exbuf, const float* den, float* outacc) {
    int t = blockIdx.x * blockDim.x + threadIdx.x;
    if (t >= 4 * 384 * 800) return;
    int c = t % 800, r = t / 800;
    int e = r % 384, b = r / 384;
    int s, d; edge_sd(esrc, edst, b, e, s, d);
    int ns = b * 128 + s, nd = b * 128 + d;
    int hd = c / 100;
    float alpha = exbuf[(b * 384 + e) * 8 + hd] / den[nd * 8 + hd];
    atomicAdd(outacc + (size_t)nd * 800 + c, alpha * Hg[(size_t)ns * 800 + c]);
}

__global__ void gat_update(const float* outacc, const float* gbias,
                           float* xg, _Float16* xgh) {
    int t = blockIdx.x * blockDim.x + threadIdx.x;
    if (t >= 512 * 800) return;
    int c = t % 800;
    float v = outacc[t] + gbias[c];
    v = v > 0.f ? v : 0.f;
    float o = xg[t] + v;
    xg[t] = o; xgh[t] = (_Float16)o;
}

// ------------- fused head GEMM: logits = gelu(P[i]+Q[j]) @ W_heads^T + b -------------
// 32x32 tile per wave: j-tile of 32 rows, n-tile of 32 cols. W stored [320,800].
__global__ void head_gemm(const float* __restrict__ P, const float* __restrict__ Q,
                          const _Float16* __restrict__ W, const float* __restrict__ bias,
                          float* __restrict__ L) {
    int lane = threadIdx.x;
    int bx = blockIdx.x;                       // 4*128*4*10 = 20480 blocks
    int nt = bx % 10, jt = (bx / 10) % 4, i = (bx / 40) % 128, b = bx / 5120;
    int n0 = nt * 32, j0 = jt * 32;
    int g = lane >> 4, lr = lane & 15;
    const float* Pr = P + (size_t)(b * 128 + i) * 800;        // includes table_b
    const float* Q0 = Q + (size_t)(b * 128 + j0 + lr) * 800;
    const float* Q1 = Q + (size_t)(b * 128 + j0 + 16 + lr) * 800;
    v8f c00 = {}, c01 = {}, c10 = {}, c11 = {};
    for (int k0 = 0; k0 < 800; k0 += 32) {
        int ko = k0 + 8 * g;
        v8f plo = *(const v8f*)(Pr + ko);
        v8f phi = *(const v8f*)(Pr + ko + 16);
        v8f q0lo = *(const v8f*)(Q0 + ko);
        v8f q0hi = *(const v8f*)(Q0 + ko + 16);
        v8f q1lo = *(const v8f*)(Q1 + ko);
        v8f q1hi = *(const v8f*)(Q1 + ko + 16);
        v16h a0, a1;
#pragma unroll
        for (int e = 0; e < 8; ++e) {
            a0[e]     = (_Float16)gelu_f(plo[e] + q0lo[e]);
            a0[e + 8] = (_Float16)gelu_f(phi[e] + q0hi[e]);
            a1[e]     = (_Float16)gelu_f(plo[e] + q1lo[e]);
            a1[e + 8] = (_Float16)gelu_f(phi[e] + q1hi[e]);
        }
        v16h b0 = load_b(W, 800, n0, k0, lane);
        v16h b1 = load_b(W, 800, n0 + 16, k0, lane);
        c00 = __builtin_amdgcn_wmma_f32_16x16x32_f16(false, a0, false, b0, (short)0, c00, false, false);
        c01 = __builtin_amdgcn_wmma_f32_16x16x32_f16(false, a0, false, b1, (short)0, c01, false, false);
        c10 = __builtin_amdgcn_wmma_f32_16x16x32_f16(false, a1, false, b0, (short)0, c10, false, false);
        c11 = __builtin_amdgcn_wmma_f32_16x16x32_f16(false, a1, false, b1, (short)0, c11, false, false);
    }
    int na = n0 + lr, nb = n0 + 16 + lr;
    float bva = bias[na], bvb = bias[nb];
    size_t rowbase = ((size_t)(b * 128 + i)) * 128;
#pragma unroll
    for (int v = 0; v < 8; ++v) {
        int ja = j0 + v + 8 * g;
        int jb = ja + 16;
        L[(rowbase + ja) * 320 + na] = c00[v] + bva;
        L[(rowbase + ja) * 320 + nb] = c01[v] + bvb;
        L[(rowbase + jb) * 320 + na] = c10[v] + bva;
        L[(rowbase + jb) * 320 + nb] = c11[v] + bvb;
    }
}

// ---------------- loss + predictions ----------------
__device__ float ls_ce(const float* l, int C, int tgt) {
    float mx = l[0];
    for (int c = 1; c < C; ++c) mx = fmaxf(mx, l[c]);
    float se = 0.f, sum = 0.f;
    for (int c = 0; c < C; ++c) { se += expf(l[c] - mx); sum += l[c]; }
    float lse = mx + logf(se);
    float smooth = (float)C * lse - sum;
    float nll = lse - l[tgt];
    return smooth * (0.1f / (float)C) + 0.9f * nll;
}

__global__ void loss_kernel(const float* L, const int* startl, const int* endl,
                            float* loss_out) {
    int t = blockIdx.x * blockDim.x + threadIdx.x;
    if (t >= 65536) return;
    int j = t % 128, i = (t / 128) % 128;
    const float* base = L + (size_t)t * 320;
    float contrib;
    if (i == j) contrib = ls_ce(base, 66, startl[t]) * (1.f / 128.f);
    else contrib = (ls_ce(base + 66, 122, startl[t]) +
                    ls_ce(base + 188, 122, endl[t])) * (1.f / 16256.f);
    atomicAdd(loss_out, contrib);
}

__global__ void pred_kernel(const float* L, float* out1, float* out2) {
    int t = blockIdx.x * blockDim.x + threadIdx.x;
    if (t >= 65536) return;
    int j = t % 128, i = (t / 128) % 128, b = t / 16384;
    const float* bij = L + (size_t)t * 320;
    float r1, r2;
    if (i == j) {
        int am = 0; float bv = bij[0];
        for (int c = 1; c < 66; ++c) if (bij[c] > bv) { bv = bij[c]; am = c; }
        r1 = r2 = (float)am;
    } else {
        const float* bji = L + (((size_t)(b * 128 + j)) * 128 + i) * 320;
        int a1 = 0; float bv1 = bij[66] + bji[66];
        for (int c = 1; c < 122; ++c) {
            float v = bij[66 + c] + bji[66 + c];
            if (v > bv1) { bv1 = v; a1 = c; }
        }
        int a2 = 0; float bv2 = bij[188] + bji[188];
        for (int c = 1; c < 122; ++c) {
            float v = bij[188 + c] + bji[188 + c];
            if (v > bv2) { bv2 = v; a2 = c; }
        }
        r1 = (float)a1; r2 = (float)a2;
    }
    out1[t] = r1; out2[t] = r2;
}

// ---------------- host orchestration ----------------
extern "C" void kernel_launch(void* const* d_in, const int* in_sizes, int n_in,
                              void* d_out, int out_size, void* d_ws, size_t ws_size,
                              hipStream_t stream) {
    const float* bert      = (const float*)d_in[0];
    const int*   pos_ids   = (const int*)d_in[1];
    const int*   edge_src  = (const int*)d_in[2];
    const int*   edge_dst  = (const int*)d_in[3];
    const int*   startl    = (const int*)d_in[4];
    const int*   endl      = (const int*)d_in[5];
    const float* pos_table = (const float*)d_in[6];
    const float* Wih_f = (const float*)d_in[7];
    const float* Whh_f = (const float*)d_in[8];
    const float* bl_f  = (const float*)d_in[9];
    const float* Wih_b = (const float*)d_in[10];
    const float* Whh_b = (const float*)d_in[11];
    const float* bl_b  = (const float*)d_in[12];
    const float* fcW = (const float*)d_in[13];
    const float* fcb = (const float*)d_in[14];
    const float* tabW = (const float*)d_in[15];
    const float* tabb = (const float*)d_in[16];
    const float* gatW = (const float*)d_in[17];
    const float* gasrc = (const float*)d_in[18];
    const float* gadst = (const float*)d_in[19];
    const float* gbias = (const float*)d_in[20];
    const float* trigW = (const float*)d_in[21];
    const float* trigb = (const float*)d_in[22];
    const float* rsW = (const float*)d_in[23];
    const float* rsb = (const float*)d_in[24];
    const float* reW = (const float*)d_in[25];
    const float* reb = (const float*)d_in[26];

    char* w = (char*)d_ws;
    size_t off = 0;
    auto alloc = [&](size_t bytes) -> void* {
        off = (off + 255) & ~(size_t)255;
        void* p = w + off;
        off += bytes;
        return p;
    };
    // padded row counts: 544 = 17 tiles of 32 (M=516); h state padded to 16 rows
    float*     xf        = (float*)alloc(516ull * 800 * 4);
    _Float16*  xh        = (_Float16*)alloc(544ull * 800 * 2);
    _Float16*  WihF_h    = (_Float16*)alloc(3200ull * 800 * 2);
    _Float16*  WhhF_h    = (_Float16*)alloc(3200ull * 800 * 2);
    _Float16*  WihB_h    = (_Float16*)alloc(3200ull * 800 * 2);
    _Float16*  WhhB_h    = (_Float16*)alloc(3200ull * 800 * 2);
    _Float16*  fcW_h     = (_Float16*)alloc(800ull * 1600 * 2);   // transposed [800,1600]
    _Float16*  tabWP_h   = (_Float16*)alloc(800ull * 800 * 2);    // transposed halves
    _Float16*  tabWQ_h   = (_Float16*)alloc(800ull * 800 * 2);
    _Float16*  gatW_h    = (_Float16*)alloc(800ull * 800 * 2);    // transposed
    _Float16*  Wheads_h  = (_Float16*)alloc(320ull * 800 * 2);    // transposed [320,800]
    float*     bheads    = (float*)alloc(320ull * 4);
    float*     XprojF    = (float*)alloc(516ull * 3200 * 4);
    float*     XprojB    = (float*)alloc(516ull * 3200 * 4);
    float*     hA_f      = (float*)alloc(16ull * 800 * 4);
    float*     hB_f      = (float*)alloc(16ull * 800 * 4);
    float*     c_f       = (float*)alloc(3200ull * 4);
    float*     hA_b      = (float*)alloc(16ull * 800 * 4);
    float*     hB_b      = (float*)alloc(16ull * 800 * 4);
    float*     c_b       = (float*)alloc(3200ull * 4);
    _Float16*  lstmo_h   = (_Float16*)alloc(544ull * 1600 * 2);
    float*     fcout     = (float*)alloc(516ull * 800 * 4);
    float*     xg        = (float*)alloc(512ull * 800 * 4);
    _Float16*  xgh       = (_Float16*)alloc(512ull * 800 * 2);
    float*     Hg        = (float*)alloc(512ull * 800 * 4);
    float*     a_s       = (float*)alloc(4096ull * 4);
    float*     a_d       = (float*)alloc(4096ull * 4);
    unsigned*  menc      = (unsigned*)alloc(4096ull * 4);
    float*     den       = (float*)alloc(4096ull * 4);
    float*     exbuf     = (float*)alloc(12288ull * 4);
    float*     outacc    = (float*)alloc(512ull * 800 * 4);
    float*     Pbuf      = (float*)alloc(512ull * 800 * 4);
    float*     Qbuf      = (float*)alloc(512ull * 800 * 4);
    float*     logits    = (float*)alloc(65536ull * 320 * 4);     // ~84 MB
    (void)ws_size; (void)in_sizes; (void)n_in; (void)out_size;

    float* dout = (float*)d_out;

    // 1) embeddings + f16 weight conversion (transposing K-major weights)
    build_x<<<(516 * 800 + TPB - 1) / TPB, TPB, 0, stream>>>(bert, pos_ids, pos_table, xf, xh);
    f2h<<<(3200 * 800 + TPB - 1) / TPB, TPB, 0, stream>>>(Wih_f, WihF_h, 3200 * 800);
    f2h<<<(3200 * 800 + TPB - 1) / TPB, TPB, 0, stream>>>(Whh_f, WhhF_h, 3200 * 800);
    f2h<<<(3200 * 800 + TPB - 1) / TPB, TPB, 0, stream>>>(Wih_b, WihB_h, 3200 * 800);
    f2h<<<(3200 * 800 + TPB - 1) / TPB, TPB, 0, stream>>>(Whh_b, WhhB_h, 3200 * 800);
    f2h_t<<<(1600 * 800 + TPB - 1) / TPB, TPB, 0, stream>>>(fcW, fcW_h, 1600, 800);
    f2h_t<<<(800 * 800 + TPB - 1) / TPB, TPB, 0, stream>>>(tabW, tabWP_h, 800, 800);
    f2h_t<<<(800 * 800 + TPB - 1) / TPB, TPB, 0, stream>>>(tabW + 800ull * 800, tabWQ_h, 800, 800);
    f2h_t<<<(800 * 800 + TPB - 1) / TPB, TPB, 0, stream>>>(gatW, gatW_h, 800, 800);
    build_heads<<<(320 * 800 + TPB - 1) / TPB, TPB, 0, stream>>>(trigW, trigb, rsW, rsb, reW, reb,
                                                                 Wheads_h, bheads);

    // 2) BiLSTM input projections (bias folded in): Xproj = x @ Wih^T + bl
    gemm_f16<<<dim3(100, 17), 32, 0, stream>>>(xh, WihF_h, bl_f, XprojF,
                                               516, 800, 800, 800, 3200);
    gemm_f16<<<dim3(100, 17), 32, 0, stream>>>(xh, WihB_h, bl_b, XprojB,
                                               516, 800, 800, 800, 3200);

    // 3) recurrence (double-buffered h; pad rows zeroed once per call)
    zero_f<<<(16 * 800 + TPB - 1) / TPB, TPB, 0, stream>>>(hA_f, 16 * 800);
    zero_f<<<(16 * 800 + TPB - 1) / TPB, TPB, 0, stream>>>(hB_f, 16 * 800);
    zero_f<<<(16 * 800 + TPB - 1) / TPB, TPB, 0, stream>>>(hA_b, 16 * 800);
    zero_f<<<(16 * 800 + TPB - 1) / TPB, TPB, 0, stream>>>(hB_b, 16 * 800);
    zero_f<<<(3200 + TPB - 1) / TPB, TPB, 0, stream>>>(c_f, 3200);
    zero_f<<<(3200 + TPB - 1) / TPB, TPB, 0, stream>>>(c_b, 3200);
    for (int s = 0; s < 129; ++s) {
        float* hpF = (s & 1) ? hB_f : hA_f; float* hnF = (s & 1) ? hA_f : hB_f;
        float* hpB = (s & 1) ? hB_b : hA_b; float* hnB = (s & 1) ? hA_b : hB_b;
        lstm_step<<<50, 32, 0, stream>>>(XprojF, WhhF_h, hpF, hnF, c_f, lstmo_h, s, 0);
        lstm_step<<<50, 32, 0, stream>>>(XprojB, WhhB_h, hpB, hnB, c_b, lstmo_h, 128 - s, 800);
    }

    // 4) lstm_fc + residual, drop CLS token
    gemm_f16<<<dim3(25, 17), 32, 0, stream>>>(lstmo_h, fcW_h, fcb, fcout,
                                              516, 1600, 1600, 1600, 800);
    fc_combine<<<(512 * 800 + TPB - 1) / TPB, TPB, 0, stream>>>(xf, fcout, xg, xgh);

    // 5) two GAT layers (shared weights)
    for (int layer = 0; layer < 2; ++layer) {
        gemm_f16<<<dim3(25, 16), 32, 0, stream>>>(xgh, gatW_h, nullptr, Hg,
                                                  512, 800, 800, 800, 800);
        att_prep<<<(4096 + TPB - 1) / TPB, TPB, 0, stream>>>(Hg, gasrc, gadst, a_s, a_d, menc, den);
        zero_f<<<(512 * 800 + TPB - 1) / TPB, TPB, 0, stream>>>(outacc, 512 * 800);
        edge_max<<<(12288 + TPB - 1) / TPB, TPB, 0, stream>>>(edge_src, edge_dst, a_s, a_d, menc);
        edge_exp<<<(12288 + TPB - 1) / TPB, TPB, 0, stream>>>(edge_src, edge_dst, a_s, a_d,
                                                              menc, den, exbuf);
        edge_aggr<<<(4 * 384 * 800 + TPB - 1) / TPB, TPB, 0, stream>>>(edge_src, edge_dst, Hg,
                                                                       exbuf, den, outacc);
        gat_update<<<(512 * 800 + TPB - 1) / TPB, TPB, 0, stream>>>(outacc, gbias, xg, xgh);
    }

    // 6) table factorization: P = x@tabW[:800]+tabb ; Q = x@tabW[800:]
    gemm_f16<<<dim3(25, 16), 32, 0, stream>>>(xgh, tabWP_h, tabb, Pbuf,
                                              512, 800, 800, 800, 800);
    gemm_f16<<<dim3(25, 16), 32, 0, stream>>>(xgh, tabWQ_h, nullptr, Qbuf,
                                              512, 800, 800, 800, 800);

    // 7) fused gelu(P+Q) @ heads  (dominant GEMM, 65536x800x320)
    head_gemm<<<20480, 32, 0, stream>>>(Pbuf, Qbuf, Wheads_h, bheads, logits);

    // 8) loss + predictions
    zero_f<<<1, 1, 0, stream>>>(dout, 1);
    loss_kernel<<<65536 / TPB, TPB, 0, stream>>>(logits, startl, endl, dout);
    pred_kernel<<<65536 / TPB, TPB, 0, stream>>>(logits, dout + 1, dout + 1 + 65536);
}
